// SparseMOE_71459665871612
// MI455X (gfx1250) — matrix operand order repaired and checked by reading, not compiled, not run on hardware
//
#include <hip/hip_runtime.h>
#include <hip/hip_bf16.h>
#include <math.h>

typedef __attribute__((ext_vector_type(16))) __bf16 v16bf;
typedef __attribute__((ext_vector_type(8)))  float  v8f;

#define H_DIM 1024
#define F_DIM 4096
#define E_NUM 8
#define N_TOK 2048
#define KT 32      // K-slab per iteration
#define MT 128     // token rows per block tile
#define NT 64      // output cols per block tile

// ---- ws byte layout ----
#define OFF_COUNTS  0u
#define OFF_OFFSETS 64u
#define OFF_BIN_TOK 128u                       // E*N ints  = 64 KB
#define OFF_BIN_W   (128u + 65536u)            // E*N floats = 64 KB
#define OFF_ABUF    (1u << 20)                 // 4096 rows * F bf16 = 32 MB

// A-matrix (16-bit, 16x32) lane mapping (ISA 7.12.2):
//   lane L: m=L%16, h=L/16 ; frag elem j -> K = j + 8*h + 8*(j>=8)
//   inverse: h=(k>>3)&1 ; j=(k&7)+((k>>4)<<3)
// B-matrix (16-bit, 32x16): lane L: n=L%16, h=L/16 ; same K order as A:
//   h=(k>>3)&1 ; j=(k&7)+((k>>4)<<3)

#define WMMA_BF16(A, B, C) \
  __builtin_amdgcn_wmma_f32_16x16x32_bf16(false, (A), false, (B), (short)0, (C), false, false)

#if defined(__has_builtin)
#if __has_builtin(__builtin_amdgcn_sched_barrier)
#define SCHED_FENCE() __builtin_amdgcn_sched_barrier(0)
#endif
#if __has_builtin(__builtin_amdgcn_global_load_async_to_lds_b128) && \
    __has_builtin(__builtin_amdgcn_s_wait_asynccnt)
#define USE_ASYNC_LDS 1
#endif
#endif
#ifndef SCHED_FENCE
#define SCHED_FENCE()
#endif
#ifndef USE_ASYNC_LDS
#define USE_ASYNC_LDS 0
#endif

#if USE_ASYNC_LDS
// Builtin parameter type per hipcc diagnostic: 'int __vector_size__(16) __device__ *'
typedef int async_i4 __attribute__((vector_size(16)));
#define AS1 __attribute__((address_space(1)))
#define AS3 __attribute__((address_space(3)))
#endif

// ---------------- zero init ----------------
__global__ __launch_bounds__(256) void moe_zero_kernel(float* __restrict__ out,
                                                       int* __restrict__ counts) {
  size_t base = (size_t)blockIdx.x * 256 * 8 + threadIdx.x;
  #pragma unroll
  for (int i = 0; i < 8; ++i) {
    size_t idx = base + (size_t)i * 256;
    if (idx < (size_t)N_TOK * H_DIM) out[idx] = 0.f;
  }
  if (blockIdx.x == 0 && threadIdx.x < E_NUM) counts[threadIdx.x] = 0;
}

// ---------------- router ----------------
__global__ __launch_bounds__(256) void moe_router_kernel(
    const float* __restrict__ x, const float* __restrict__ noise,
    const float* __restrict__ Wg, const float* __restrict__ bg,
    float* __restrict__ out, int* __restrict__ counts,
    int* __restrict__ bin_tok, float* __restrict__ bin_w) {
  const int n = blockIdx.x;
  const int tid = threadIdx.x;
  float p[E_NUM];
  #pragma unroll
  for (int e = 0; e < E_NUM; ++e) p[e] = 0.f;
  for (int h = tid; h < H_DIM; h += 256) {
    float xv = x[(size_t)n * H_DIM + h];
    #pragma unroll
    for (int e = 0; e < E_NUM; ++e) p[e] += xv * Wg[h * E_NUM + e];
  }
  __shared__ float red[256][E_NUM];
  #pragma unroll
  for (int e = 0; e < E_NUM; ++e) red[tid][e] = p[e];
  __syncthreads();
  for (int s = 128; s > 0; s >>= 1) {
    if (tid < s) {
      #pragma unroll
      for (int e = 0; e < E_NUM; ++e) red[tid][e] += red[tid + s][e];
    }
    __syncthreads();
  }
  if (tid == 0) {
    float lg[E_NUM];
    float* logits_out = out + (size_t)N_TOK * H_DIM;
    #pragma unroll
    for (int e = 0; e < E_NUM; ++e) {
      lg[e] = red[0][e] + bg[e] + 0.01f * noise[n * E_NUM + e];
      logits_out[(size_t)n * E_NUM + e] = lg[e];
    }
    float mx = lg[0];
    #pragma unroll
    for (int e = 1; e < E_NUM; ++e) mx = fmaxf(mx, lg[e]);
    float ex[E_NUM], ssum = 0.f;
    #pragma unroll
    for (int e = 0; e < E_NUM; ++e) { ex[e] = __expf(lg[e] - mx); ssum += ex[e]; }
    int e1 = 0;
    #pragma unroll
    for (int e = 1; e < E_NUM; ++e) if (lg[e] > lg[e1]) e1 = e;
    int e2 = (e1 == 0) ? 1 : 0;
    #pragma unroll
    for (int e = 0; e < E_NUM; ++e) if (e != e1 && lg[e] > lg[e2]) e2 = e;
    float p1 = ex[e1] / ssum, p2 = ex[e2] / ssum;
    float inv = 1.f / (p1 + p2);
    int* sel = (int*)(out + (size_t)N_TOK * H_DIM + (size_t)N_TOK * E_NUM);
    sel[n * 2 + 0] = e1;
    sel[n * 2 + 1] = e2;
    int pos1 = atomicAdd(&counts[e1], 1);
    bin_tok[e1 * N_TOK + pos1] = n;
    bin_w[e1 * N_TOK + pos1] = p1 * inv;
    int pos2 = atomicAdd(&counts[e2], 1);
    bin_tok[e2 * N_TOK + pos2] = n;
    bin_w[e2 * N_TOK + pos2] = p2 * inv;
  }
}

// ---------------- prefix offsets ----------------
__global__ void moe_offsets_kernel(const int* __restrict__ counts, int* __restrict__ offsets) {
  if (threadIdx.x == 0 && blockIdx.x == 0) {
    int run = 0;
    for (int e = 0; e < E_NUM; ++e) { offsets[e] = run; run += counts[e]; }
  }
}

// ---------------- FFN stage 1: act = silu(x W1 + b1) * (x W3 + b3) ----------------
__global__ __launch_bounds__(256) void moe_ffn1_kernel(
    const float* __restrict__ x,
    const float* __restrict__ W1, const float* __restrict__ b1,
    const float* __restrict__ W3, const float* __restrict__ b3,
    const int* __restrict__ counts, const int* __restrict__ offsets,
    const int* __restrict__ bin_tok, __bf16* __restrict__ Abuf) {
  const int e = blockIdx.z;
  const int tile = blockIdx.y;
  const int cnt = counts[e];
  if (tile * MT >= cnt) return;
  const int fbase = blockIdx.x * NT;
  const int tid = threadIdx.x;
  const int lane = tid & 31;
  const int wv = tid >> 5;

  __shared__ __align__(32) __bf16 lsX[2][MT * KT];      // 2 x 8 KB
  __shared__ __align__(32) __bf16 lsW[2][2 * KT * NT];  // 2 x 8 KB
  __shared__ int toks[MT];

  for (int i = tid; i < MT; i += 256) {
    int r = tile * MT + i;
    toks[i] = (r < cnt) ? bin_tok[e * N_TOK + r] : -1;
  }
  __syncthreads();

  const float* W1e = W1 + (size_t)e * H_DIM * F_DIM;
  const float* W3e = W3 + (size_t)e * H_DIM * F_DIM;

  // ---- loop-invariant loader geometry (4 float4-quads of X, 4 of W per thread) ----
  int xlds[4]; const float* xp[4]; bool xv[4];
  int wlds[4]; const float* wp[4];
  #pragma unroll
  for (int qi = 0; qi < 4; ++qi) {
    int qid = tid + 256 * qi;            // 0..1023 quads
    // X: MT x KT, 8 quads per row
    int row = qid >> 3;
    int k0 = (qid & 7) * 4;
    int hx = (k0 >> 3) & 1;
    int jx = (k0 & 7) + ((k0 >> 4) << 3);
    xlds[qi] = ((row >> 4) * 2 + hx) * 256 + (row & 15) * 16 + jx;
    int t = toks[row];
    xv[qi] = (t >= 0);
    xp[qi] = x + (size_t)(t < 0 ? 0 : t) * H_DIM + k0;
    // W: 2 mats x KT x NT, 16 quads per k-row
    int mat = qid >> 9;
    int r = qid & 511;
    int k = r >> 4;
    int f0 = (r & 15) * 4;
    int hw = (k >> 3) & 1, jw = (k & 7) + ((k >> 4) << 3);
    wlds[qi] = ((mat * 4 + (f0 >> 4)) * 2 + hw) * 256 + (f0 & 15) * 16 + jw;
    wp[qi] = (mat ? W3e : W1e) + (size_t)k * F_DIM + fbase + f0;
  }

  float4 xr[4], wr[4];
  auto stage_load = [&](int kb) {
    #pragma unroll
    for (int qi = 0; qi < 4; ++qi) {
      xr[qi] = xv[qi] ? *(const float4*)(xp[qi] + kb) : make_float4(0.f, 0.f, 0.f, 0.f);
      wr[qi] = *(const float4*)(wp[qi] + (size_t)kb * F_DIM);
    }
  };
  auto stage_store = [&](int buf) {
    #pragma unroll
    for (int qi = 0; qi < 4; ++qi) {
      __bf16* px = &lsX[buf][xlds[qi]];   // 4 contiguous shorts
      px[0] = (__bf16)xr[qi].x; px[1] = (__bf16)xr[qi].y;
      px[2] = (__bf16)xr[qi].z; px[3] = (__bf16)xr[qi].w;
      __bf16* pw = &lsW[buf][wlds[qi]];   // stride-16 scatter (transpose)
      pw[0]  = (__bf16)wr[qi].x; pw[16] = (__bf16)wr[qi].y;
      pw[32] = (__bf16)wr[qi].z; pw[48] = (__bf16)wr[qi].w;
    }
  };

  v8f acc_h[4] = {};
  v8f acc_g[4] = {};

  int buf = 0;
  stage_load(0);
  stage_store(0);
  __syncthreads();

  for (int kb = 0; kb < H_DIM; kb += KT) {
    const bool has_next = (kb + KT) < H_DIM;
    if (has_next) stage_load(kb + KT);    // global loads overlap WMMA below
    // Load ALL fragments first, then fence the scheduler so the WMMA chain is
    // not interleaved with (and serialized on) its own ds_loads.
    v16bf a = *(const v16bf*)(lsX[buf] + wv * 512 + lane * 16);
    v16bf bw[8];
    #pragma unroll
    for (int ng = 0; ng < 4; ++ng) {
      bw[ng]     = *(const v16bf*)(lsW[buf] + ng * 512 + lane * 16);
      bw[ng + 4] = *(const v16bf*)(lsW[buf] + 2048 + ng * 512 + lane * 16);
    }
    SCHED_FENCE();
    #pragma unroll
    for (int ng = 0; ng < 4; ++ng) {
      acc_h[ng] = WMMA_BF16(a, bw[ng],     acc_h[ng]);
      acc_g[ng] = WMMA_BF16(a, bw[ng + 4], acc_g[ng]);
    }
    if (has_next) stage_store(buf ^ 1);   // safe: other buffer; prev reads fenced by last barrier
    __syncthreads();
    buf ^= 1;
  }

  // epilogue: silu(h)*g -> Abuf (bf16), rows compacted at offsets[e]
  const int aoff = offsets[e];
  const int halfsel = lane >> 4;
  const int ncl = lane & 15;
  #pragma unroll
  for (int ng = 0; ng < 4; ++ng) {
    int fc = fbase + ng * 16 + ncl;
    float bb1 = b1[e * F_DIM + fc];
    float bb3 = b3[e * F_DIM + fc];
    #pragma unroll
    for (int r = 0; r < 8; ++r) {
      int mrow = wv * 16 + r + 8 * halfsel;
      int rglob = tile * MT + mrow;
      if (rglob < cnt) {
        float hv = acc_h[ng][r] + bb1;
        float gv = acc_g[ng][r] + bb3;
        float act = (hv / (1.f + __expf(-hv))) * gv;
        Abuf[(size_t)(aoff + rglob) * F_DIM + fc] = (__bf16)act;
      }
    }
  }
}

// ---------------- FFN stage 2: y = act W2 + b2, weighted scatter-add ----------------
__global__ __launch_bounds__(256) void moe_ffn2_kernel(
    const __bf16* __restrict__ Abuf,
    const float* __restrict__ W2, const float* __restrict__ b2,
    const int* __restrict__ counts, const int* __restrict__ offsets,
    const int* __restrict__ bin_tok, const float* __restrict__ bin_w,
    float* __restrict__ out) {
  const int e = blockIdx.z;
  const int tile = blockIdx.y;
  const int cnt = counts[e];
  if (tile * MT >= cnt) return;
  const int hbase = blockIdx.x * NT;
  const int tid = threadIdx.x;
  const int lane = tid & 31;
  const int wv = tid >> 5;

  __shared__ __align__(32) __bf16 lsA[2][MT * KT];   // 2 x 8 KB
  __shared__ __align__(32) __bf16 lsW2[2][KT * NT];  // 2 x 4 KB

  const int aoff = offsets[e];
  const float* W2e = W2 + (size_t)e * F_DIM * H_DIM;

  // ---- loop-invariant loader geometry ----
  // A: 2 octs (8 bf16 = 16 B) per thread; 4 octs per row (already bf16 -> async copy)
  int alds[2]; const __bf16* ap[2]; bool av[2];
  // W2: 2 float4-quads per thread; 16 quads per k-row
  int w2lds[2]; const float* w2p[2];
  #pragma unroll
  for (int i = 0; i < 2; ++i) {
    int aid = tid + 256 * i;             // 0..511 octs
    int row = aid >> 2;
    int k0 = (aid & 3) * 8;
    int ha = (k0 >> 3) & 1;
    int ja = (k0 >> 4) << 3;             // k0 % 8 == 0
    alds[i] = ((row >> 4) * 2 + ha) * 256 + (row & 15) * 16 + ja;
    int rglob = tile * MT + row;
    av[i] = (rglob < cnt);
    // invalid rows alias row 0 of this expert: harmless, epilogue masks them
    ap[i] = Abuf + (size_t)(aoff + (av[i] ? rglob : 0)) * F_DIM + k0;

    int qid = tid + 256 * i;             // 0..511 quads
    int k = qid >> 4;
    int f0 = (qid & 15) * 4;
    int hw = (k >> 3) & 1, jw = (k & 7) + ((k >> 4) << 3);
    w2lds[i] = ((f0 >> 4) * 2 + hw) * 256 + (f0 & 15) * 16 + jw;
    w2p[i] = W2e + (size_t)k * H_DIM + hbase + f0;
  }

#if !USE_ASYNC_LDS
  uint4 ar[2];
#endif
  float4 w2r[2];

  auto a_issue = [&](int kb, int dbuf) {
#if USE_ASYNC_LDS
    #pragma unroll
    for (int i = 0; i < 2; ++i)
      __builtin_amdgcn_global_load_async_to_lds_b128(
          (AS1 async_i4*)(ap[i] + kb),
          (AS3 async_i4*)&lsA[dbuf][alds[i]],
          0, 0);
#else
    (void)dbuf;
    #pragma unroll
    for (int i = 0; i < 2; ++i)
      ar[i] = av[i] ? *(const uint4*)(ap[i] + kb) : make_uint4(0u, 0u, 0u, 0u);
#endif
  };
  auto w2_load = [&](int kb) {
    #pragma unroll
    for (int i = 0; i < 2; ++i)
      w2r[i] = *(const float4*)(w2p[i] + (size_t)kb * H_DIM);
  };
  auto stage_store = [&](int dbuf) {
    #pragma unroll
    for (int i = 0; i < 2; ++i) {
#if !USE_ASYNC_LDS
      *(uint4*)&lsA[dbuf][alds[i]] = ar[i];          // ds_store_b128
#endif
      __bf16* pw = &lsW2[dbuf][w2lds[i]];
      pw[0]  = (__bf16)w2r[i].x; pw[16] = (__bf16)w2r[i].y;
      pw[32] = (__bf16)w2r[i].z; pw[48] = (__bf16)w2r[i].w;
    }
  };

  v8f acc[4] = {};

  int buf = 0;
  a_issue(0, 0);
  w2_load(0);
  stage_store(0);
#if USE_ASYNC_LDS
  __builtin_amdgcn_s_wait_asynccnt(0);
#endif
  __syncthreads();

  for (int kb = 0; kb < F_DIM; kb += KT) {
    const bool has_next = (kb + KT) < F_DIM;
    if (has_next) {
      a_issue(kb + KT, buf ^ 1);   // async DMA into other buffer, overlaps WMMA
      w2_load(kb + KT);
    }
    v16bf a = *(const v16bf*)(lsA[buf] + wv * 512 + lane * 16);
    v16bf bw[4];
    #pragma unroll
    for (int ng = 0; ng < 4; ++ng)
      bw[ng] = *(const v16bf*)(lsW2[buf] + ng * 512 + lane * 16);
    SCHED_FENCE();
    #pragma unroll
    for (int ng = 0; ng < 4; ++ng)
      acc[ng] = WMMA_BF16(a, bw[ng], acc[ng]);
    if (has_next) {
      stage_store(buf ^ 1);
#if USE_ASYNC_LDS
      __builtin_amdgcn_s_wait_asynccnt(0);
#endif
    }
    __syncthreads();
    buf ^= 1;
  }

  const int halfsel = lane >> 4;
  const int ncl = lane & 15;
  #pragma unroll
  for (int ng = 0; ng < 4; ++ng) {
    int hc = hbase + ng * 16 + ncl;
    float bb2 = b2[e * H_DIM + hc];
    #pragma unroll
    for (int r = 0; r < 8; ++r) {
      int mrow = wv * 16 + r + 8 * halfsel;
      int rglob = tile * MT + mrow;
      if (rglob < cnt) {
        int t = bin_tok[e * N_TOK + rglob];
        float wgt = bin_w[e * N_TOK + rglob];
        float val = (acc[ng][r] + bb2) * wgt;
        atomicAdd(&out[(size_t)t * H_DIM + hc], val);
      }
    }
  }
}

extern "C" void kernel_launch(void* const* d_in, const int* in_sizes, int n_in,
                              void* d_out, int out_size, void* d_ws, size_t ws_size,
                              hipStream_t stream) {
  const float* x     = (const float*)d_in[0];
  const float* noise = (const float*)d_in[1];
  const float* Wg    = (const float*)d_in[2];
  const float* bg    = (const float*)d_in[3];
  const float* W1    = (const float*)d_in[4];
  const float* b1    = (const float*)d_in[5];
  const float* W2    = (const float*)d_in[6];
  const float* b2    = (const float*)d_in[7];
  const float* W3    = (const float*)d_in[8];
  const float* b3    = (const float*)d_in[9];
  float* out = (float*)d_out;

  char* ws = (char*)d_ws;
  int*    counts  = (int*)(ws + OFF_COUNTS);
  int*    offsets = (int*)(ws + OFF_OFFSETS);
  int*    bin_tok = (int*)(ws + OFF_BIN_TOK);
  float*  bin_w   = (float*)(ws + OFF_BIN_W);
  __bf16* Abuf    = (__bf16*)(ws + OFF_ABUF);

  moe_zero_kernel<<<dim3((N_TOK * H_DIM) / (256 * 8)), dim3(256), 0, stream>>>(out, counts);
  moe_router_kernel<<<dim3(N_TOK), dim3(256), 0, stream>>>(
      x, noise, Wg, bg, out, counts, bin_tok, bin_w);
  moe_offsets_kernel<<<dim3(1), dim3(32), 0, stream>>>(counts, offsets);
  moe_ffn1_kernel<<<dim3(F_DIM / NT, N_TOK / MT, E_NUM), dim3(256), 0, stream>>>(
      x, W1, b1, W3, b3, counts, offsets, bin_tok, Abuf);
  moe_ffn2_kernel<<<dim3(H_DIM / NT, N_TOK / MT, E_NUM), dim3(256), 0, stream>>>(
      Abuf, W2, b2, counts, offsets, bin_tok, bin_w, out);
}
